// GATv2_32427003084907
// MI455X (gfx1250) — compile-verified, matrix operand order checked by reading, not run on hardware
//
#include <hip/hip_runtime.h>
#include <hip/hip_bf16.h>
#include <math.h>

// ---------------------------------------------------------------------------
// GATv2 x2 layers for MI455X (gfx1250, wave32).
// Dense projections via v_wmma_f32_16x16x32_f16 with pre-converted f16
// operands (A: row-major f16, B: transposed f16) so every fragment is two
// 16-byte b128 loads per lane. Edge phase: float4-vectorized gathers and
// native f32 atomics (entire working set is L2-resident: 192MB L2).
// ---------------------------------------------------------------------------

typedef __attribute__((ext_vector_type(16))) _Float16 v16h;
typedef __attribute__((ext_vector_type(8)))  _Float16 v8h;
typedef __attribute__((ext_vector_type(4)))  _Float16 v4h;
typedef __attribute__((ext_vector_type(8)))  float    v8f;
typedef __attribute__((ext_vector_type(4)))  float    v4f;

#define SLOPE 0.2f

// float atomic-max via signed/unsigned integer atomics (no CAS loop).
__device__ __forceinline__ void atomicMaxF32(float* addr, float val) {
    if (val >= 0.0f) {
        atomicMax((int*)addr, __float_as_int(val));
    } else {
        atomicMin((unsigned int*)addr, (unsigned int)__float_as_int(val));
    }
}

// ---------------------------------------------------------------------------
// Prep: x (f32 [N,128]) -> xh (f16 [N,128]); 4 elements per thread.
// ---------------------------------------------------------------------------
__global__ void k_cvt_x(const float* __restrict__ x, _Float16* __restrict__ xh,
                        int N) {
    const int idx = blockIdx.x * blockDim.x + threadIdx.x;
    if (idx >= N * 32) return;
    const v4f v = *(const v4f*)(x + (size_t)idx * 4);
    v4h h;
    h[0] = (_Float16)v[0]; h[1] = (_Float16)v[1];
    h[2] = (_Float16)v[2]; h[3] = (_Float16)v[3];
    *(v4h*)(xh + (size_t)idx * 4) = h;
}

// Prep: Wl1/Wr1 (f32 [128,64]) -> transposed f16 [64,128].
__global__ void k_cvt_w1(const float* __restrict__ Wl, const float* __restrict__ Wr,
                         _Float16* __restrict__ Wlt, _Float16* __restrict__ Wrt) {
    const int idx = blockIdx.x * blockDim.x + threadIdx.x;
    if (idx >= 64 * 128) return;
    const int n = idx >> 7;
    const int k = idx & 127;
    Wlt[idx] = (_Float16)Wl[k * 64 + n];
    Wrt[idx] = (_Float16)Wr[k * 64 + n];
}

// Prep: Wl2/Wr2 (f32 [64,16]) -> transposed f16 [16,64].
__global__ void k_cvt_w2(const float* __restrict__ Wl, const float* __restrict__ Wr,
                         _Float16* __restrict__ Wlt, _Float16* __restrict__ Wrt) {
    const int idx = blockIdx.x * blockDim.x + threadIdx.x;
    if (idx >= 16 * 64) return;
    const int n = idx >> 6;
    const int k = idx & 63;
    Wlt[idx] = (_Float16)Wl[k * 16 + n];
    Wrt[idx] = (_Float16)Wr[k * 16 + n];
}

// ---------------------------------------------------------------------------
// Init for layer 1: agg1 = 0 (N*64), amax = -inf (N*2), denom = 0 (N*2),
// d_out = 0 (N*16).
// ---------------------------------------------------------------------------
__global__ void k_init1(float* __restrict__ agg1, float* __restrict__ amax,
                        float* __restrict__ denom, float* __restrict__ out,
                        int N) {
    int idx = blockIdx.x * blockDim.x + threadIdx.x;
    if (idx >= N * 64) return;
    agg1[idx] = 0.0f;
    if (idx < N * 2) { amax[idx] = -__builtin_inff(); denom[idx] = 0.0f; }
    if (idx < N * 16) out[idx] = 0.0f;
}

// Init for layer 2: amax = -inf (N), denom = 0 (N).
__global__ void k_init2(float* __restrict__ amax, float* __restrict__ denom,
                        int N) {
    int idx = blockIdx.x * blockDim.x + threadIdx.x;
    if (idx >= N) return;
    amax[idx] = -__builtin_inff();
    denom[idx] = 0.0f;
}

// ---------------------------------------------------------------------------
// Fragment builders. 16-bit A 16x32 layout: lane group g = lane>>4 holds
// K in [g*8, g*8+8) and [16+g*8, 16+g*8+8): two contiguous runs of 8 halves.
// ---------------------------------------------------------------------------
__device__ __forceinline__ v16h load_frag_k32(const _Float16* __restrict__ rowp,
                                              int kb, int kgrp) {
    const int k0 = kb + kgrp * 8;
    const int k1 = kb + 16 + kgrp * 8;
    const v8h lo = *(const v8h*)(rowp + k0);   // 16B aligned
    const v8h hi = *(const v8h*)(rowp + k1);
    v16h f;
    #pragma unroll
    for (int i = 0; i < 8; ++i) { f[i] = lo[i]; f[i + 8] = hi[i]; }
    return f;
}

// ---------------------------------------------------------------------------
// GEMM1: xl = x @ Wl1, xr = x @ Wr1.  xh:[N,128] f16, Wt:[64,128] f16 (W^T).
// block = 64 threads (2 waves): wave0 -> Wl path, wave1 -> Wr path.
// One 16-row tile per block; 4 column tiles (HC=64); K loop 128 step 32.
// ---------------------------------------------------------------------------
__global__ void k_gemm1(const _Float16* __restrict__ xh,
                        const _Float16* __restrict__ Wlt,
                        const _Float16* __restrict__ Wrt,
                        float* __restrict__ xl,
                        float* __restrict__ xr,
                        int N) {
    const int lane = threadIdx.x & 31;
    const int wave = threadIdx.x >> 5;
    const _Float16* __restrict__ Wt  = wave ? Wrt : Wlt;
    float* __restrict__          out = wave ? xr : xl;

    const int row0 = blockIdx.x * 16;
    const int rA   = lane & 15;
    const int kgrp = lane >> 4;
    const int row  = row0 + rA;
    const bool rowOK = row < N;
    const _Float16* __restrict__ xrow = xh + (size_t)(rowOK ? row : 0) * 128;

    v8f acc[4] = {v8f{}, v8f{}, v8f{}, v8f{}};

    #pragma unroll
    for (int kb = 0; kb < 128; kb += 32) {
        const v16h a = load_frag_k32(xrow, kb, kgrp);
        #pragma unroll
        for (int nt = 0; nt < 4; ++nt) {
            const _Float16* __restrict__ wrow = Wt + (size_t)(nt * 16 + rA) * 128;
            const v16h b = load_frag_k32(wrow, kb, kgrp);
            acc[nt] = __builtin_amdgcn_wmma_f32_16x16x32_f16(
                false, a, false, b, (short)0, acc[nt], false, false);
        }
    }

    // C/D layout: VGPR r, lane l -> row = r + 8*(l>>4), col = l&15.
    #pragma unroll
    for (int nt = 0; nt < 4; ++nt) {
        #pragma unroll
        for (int r = 0; r < 8; ++r) {
            const int orow = row0 + r + 8 * kgrp;
            if (orow < N) out[(size_t)orow * 64 + nt * 16 + rA] = acc[nt][r];
        }
    }
}

// ---------------------------------------------------------------------------
// GEMM2: xl2 = h @ Wl2, xr2 = h @ Wr2.  hh:[N,64] f16, Wt:[16,64] f16 (W^T).
// ---------------------------------------------------------------------------
__global__ void k_gemm2(const _Float16* __restrict__ hh,
                        const _Float16* __restrict__ Wlt,
                        const _Float16* __restrict__ Wrt,
                        float* __restrict__ xl2,
                        float* __restrict__ xr2,
                        int N) {
    const int lane = threadIdx.x & 31;
    const int wave = threadIdx.x >> 5;
    const _Float16* __restrict__ Wt  = wave ? Wrt : Wlt;
    float* __restrict__          out = wave ? xr2 : xl2;

    const int row0 = blockIdx.x * 16;
    const int rA   = lane & 15;
    const int kgrp = lane >> 4;
    const int row  = row0 + rA;
    const bool rowOK = row < N;
    const _Float16* __restrict__ hrow = hh + (size_t)(rowOK ? row : 0) * 64;
    const _Float16* __restrict__ wrow = Wt + (size_t)rA * 64;

    v8f acc = v8f{};
    #pragma unroll
    for (int kb = 0; kb < 64; kb += 32) {
        const v16h a = load_frag_k32(hrow, kb, kgrp);
        const v16h b = load_frag_k32(wrow, kb, kgrp);
        acc = __builtin_amdgcn_wmma_f32_16x16x32_f16(
            false, a, false, b, (short)0, acc, false, false);
    }
    #pragma unroll
    for (int r = 0; r < 8; ++r) {
        const int orow = row0 + r + 8 * kgrp;
        if (orow < N) out[(size_t)orow * 16 + rA] = acc[r];
    }
}

// ---------------------------------------------------------------------------
// Layer-1 edge logits: one thread per (edge, head); float4-vectorized.
// ---------------------------------------------------------------------------
__global__ void k_edge1_logits(const int* __restrict__ ei,
                               const float* __restrict__ xl1,
                               const float* __restrict__ xr1,
                               const float* __restrict__ eattr,
                               const float* __restrict__ We1,
                               const float* __restrict__ att1,
                               float* __restrict__ elog,
                               float* __restrict__ amax,
                               int E) {
    const int idx = blockIdx.x * blockDim.x + threadIdx.x;
    if (idx >= E * 2) return;
    const int e = idx >> 1;
    const int h = idx & 1;
    const int src = ei[e];
    const int dst = ei[E + e];

    float ea[6];
    #pragma unroll
    for (int d = 0; d < 6; ++d) ea[d] = eattr[(size_t)e * 6 + d];

    const v4f* __restrict__ xlp = (const v4f*)(xl1 + (size_t)src * 64 + h * 32);
    const v4f* __restrict__ xrp = (const v4f*)(xr1 + (size_t)dst * 64 + h * 32);
    const v4f* __restrict__ atp = (const v4f*)(att1 + h * 32);

    float lsum = 0.0f;
    #pragma unroll
    for (int q = 0; q < 8; ++q) {
        const v4f a  = xlp[q];
        const v4f bq = xrp[q];
        const v4f at = atp[q];
        v4f m;
        #pragma unroll
        for (int j = 0; j < 4; ++j) m[j] = a[j] + bq[j];
        #pragma unroll
        for (int d = 0; d < 6; ++d) {
            const v4f wq = *(const v4f*)(We1 + d * 64 + h * 32 + q * 4);
            #pragma unroll
            for (int j = 0; j < 4; ++j) m[j] += ea[d] * wq[j];
        }
        #pragma unroll
        for (int j = 0; j < 4; ++j) {
            const float lr = m[j] > 0.0f ? m[j] : SLOPE * m[j];
            lsum += lr * at[j];
        }
    }
    elog[idx] = lsum;
    atomicMaxF32(&amax[dst * 2 + h], lsum);
}

// exp(logit - amax[dst]) -> elog (in place); atomicAdd into denom[dst].
__global__ void k_edge1_expsum(const int* __restrict__ ei,
                               float* __restrict__ elog,
                               const float* __restrict__ amax,
                               float* __restrict__ denom,
                               int E) {
    const int idx = blockIdx.x * blockDim.x + threadIdx.x;
    if (idx >= E * 2) return;
    const int e = idx >> 1;
    const int h = idx & 1;
    const int dst = ei[E + e];
    const float ev = __expf(elog[idx] - amax[dst * 2 + h]);
    elog[idx] = ev;
    atomicAdd(&denom[dst * 2 + h], ev);
}

// Aggregate: one thread per (edge, 4-channel quad); 4 f32 atomics per thread.
__global__ void k_edge1_agg(const int* __restrict__ ei,
                            const float* __restrict__ xl1,
                            const float* __restrict__ elog,
                            const float* __restrict__ denom,
                            float* __restrict__ agg1,
                            int E) {
    const int idx = blockIdx.x * blockDim.x + threadIdx.x;
    if (idx >= E * 16) return;
    const int e  = idx >> 4;
    const int c4 = (idx & 15) * 4;       // quad never straddles heads
    const int h  = c4 >> 5;
    const int src = ei[e];
    const int dst = ei[E + e];
    const float alpha = elog[e * 2 + h] / (denom[dst * 2 + h] + 1e-16f);
    const v4f v = *(const v4f*)(xl1 + (size_t)src * 64 + c4);
    float* __restrict__ ap = agg1 + (size_t)dst * 64 + c4;
    #pragma unroll
    for (int j = 0; j < 4; ++j) atomicAdd(ap + j, v[j] * alpha);
}

// h = elu(agg1 + b1) -> hh (f16) for GEMM2.
__global__ void k_elu_bias(const float* __restrict__ agg1,
                           const float* __restrict__ b1,
                           _Float16* __restrict__ hh, int N) {
    const int idx = blockIdx.x * blockDim.x + threadIdx.x;
    if (idx >= N * 64) return;
    const float v = agg1[idx] + b1[idx & 63];
    hh[idx] = (_Float16)(v > 0.0f ? v : (__expf(v) - 1.0f));
}

// ---------------------------------------------------------------------------
// Layer-2 edge kernels (H=1, C=16, no edge features).
// ---------------------------------------------------------------------------
__global__ void k_edge2_logits(const int* __restrict__ ei,
                               const float* __restrict__ xl2,
                               const float* __restrict__ xr2,
                               const float* __restrict__ att2,
                               float* __restrict__ elog,
                               float* __restrict__ amax,
                               int E) {
    const int e = blockIdx.x * blockDim.x + threadIdx.x;
    if (e >= E) return;
    const int src = ei[e];
    const int dst = ei[E + e];
    const v4f* __restrict__ xlp = (const v4f*)(xl2 + (size_t)src * 16);
    const v4f* __restrict__ xrp = (const v4f*)(xr2 + (size_t)dst * 16);
    float lsum = 0.0f;
    #pragma unroll
    for (int q = 0; q < 4; ++q) {
        const v4f a  = xlp[q];
        const v4f bq = xrp[q];
        const v4f at = *(const v4f*)(att2 + q * 4);
        #pragma unroll
        for (int j = 0; j < 4; ++j) {
            const float m = a[j] + bq[j];
            const float lr = m > 0.0f ? m : SLOPE * m;
            lsum += lr * at[j];
        }
    }
    elog[e] = lsum;
    atomicMaxF32(&amax[dst], lsum);
}

__global__ void k_edge2_expsum(const int* __restrict__ ei,
                               float* __restrict__ elog,
                               const float* __restrict__ amax,
                               float* __restrict__ denom,
                               int E) {
    const int e = blockIdx.x * blockDim.x + threadIdx.x;
    if (e >= E) return;
    const int dst = ei[E + e];
    const float ev = __expf(elog[e] - amax[dst]);
    elog[e] = ev;
    atomicAdd(&denom[dst], ev);
}

// One thread per (edge, 4-channel quad).
__global__ void k_edge2_agg(const int* __restrict__ ei,
                            const float* __restrict__ xl2,
                            const float* __restrict__ elog,
                            const float* __restrict__ denom,
                            float* __restrict__ out,
                            int E) {
    const int idx = blockIdx.x * blockDim.x + threadIdx.x;
    if (idx >= E * 4) return;
    const int e  = idx >> 2;
    const int c4 = (idx & 3) * 4;
    const int src = ei[e];
    const int dst = ei[E + e];
    const float alpha = elog[e] / (denom[dst] + 1e-16f);
    const v4f v = *(const v4f*)(xl2 + (size_t)src * 16 + c4);
    float* __restrict__ op = out + (size_t)dst * 16 + c4;
    #pragma unroll
    for (int j = 0; j < 4; ++j) atomicAdd(op + j, v[j] * alpha);
}

__global__ void k_final_bias(float* __restrict__ out,
                             const float* __restrict__ b2, int N) {
    const int idx = blockIdx.x * blockDim.x + threadIdx.x;
    if (idx >= N * 16) return;
    out[idx] += b2[idx & 15];
}

// ---------------------------------------------------------------------------
// Launch
// ---------------------------------------------------------------------------
extern "C" void kernel_launch(void* const* d_in, const int* in_sizes, int n_in,
                              void* d_out, int out_size, void* d_ws, size_t ws_size,
                              hipStream_t stream) {
    const float* x     = (const float*)d_in[0];
    const int*   ei    = (const int*)  d_in[1];
    const float* eattr = (const float*)d_in[2];
    const float* Wl1   = (const float*)d_in[3];
    const float* Wr1   = (const float*)d_in[4];
    const float* We1   = (const float*)d_in[5];
    const float* att1  = (const float*)d_in[6];
    const float* b1    = (const float*)d_in[7];
    const float* Wl2   = (const float*)d_in[8];
    const float* Wr2   = (const float*)d_in[9];
    const float* att2  = (const float*)d_in[10];
    const float* b2    = (const float*)d_in[11];
    float* out = (float*)d_out;

    const int N = in_sizes[0] / 128;
    const int E = in_sizes[1] / 2;

    // Workspace carve-out, 256B-aligned regions.
    char* wsp = (char*)d_ws;
    auto alloc = [&](size_t bytes) -> void* {
        void* p = (void*)wsp;
        wsp += (bytes + 255) & ~(size_t)255;
        return p;
    };
    float* xl1   = (float*)alloc((size_t)N * 64 * 4);
    float* xr1   = (float*)alloc((size_t)N * 64 * 4);
    float* agg1  = (float*)alloc((size_t)N * 64 * 4);
    float* xl2   = (float*)alloc((size_t)N * 16 * 4);
    float* xr2   = (float*)alloc((size_t)N * 16 * 4);
    float* elog  = (float*)alloc((size_t)E * 2 * 4);
    float* amax  = (float*)alloc((size_t)N * 2 * 4);
    float* denom = (float*)alloc((size_t)N * 2 * 4);
    _Float16* xh   = (_Float16*)alloc((size_t)N * 128 * 2);
    _Float16* hh   = (_Float16*)alloc((size_t)N * 64 * 2);
    _Float16* Wl1t = (_Float16*)alloc(64 * 128 * 2);
    _Float16* Wr1t = (_Float16*)alloc(64 * 128 * 2);
    _Float16* Wl2t = (_Float16*)alloc(16 * 64 * 2);
    _Float16* Wr2t = (_Float16*)alloc(16 * 64 * 2);

    const int TB = 256;
    auto blk = [&](long long n) { return (unsigned)((n + TB - 1) / TB); };

    // ---- Prep (f16 operand images) ----
    k_cvt_x <<<blk((long long)N * 32), TB, 0, stream>>>(x, xh, N);
    k_cvt_w1<<<blk(64 * 128), TB, 0, stream>>>(Wl1, Wr1, Wl1t, Wr1t);
    k_cvt_w2<<<blk(16 * 64), TB, 0, stream>>>(Wl2, Wr2, Wl2t, Wr2t);

    // ---- Layer 1 ----
    k_init1<<<blk((long long)N * 64), TB, 0, stream>>>(agg1, amax, denom, out, N);
    k_gemm1<<<(N + 15) / 16, 64, 0, stream>>>(xh, Wl1t, Wr1t, xl1, xr1, N);
    k_edge1_logits<<<blk((long long)E * 2), TB, 0, stream>>>(
        ei, xl1, xr1, eattr, We1, att1, elog, amax, E);
    k_edge1_expsum<<<blk((long long)E * 2), TB, 0, stream>>>(ei, elog, amax, denom, E);
    k_edge1_agg<<<blk((long long)E * 16), TB, 0, stream>>>(ei, xl1, elog, denom, agg1, E);
    k_elu_bias<<<blk((long long)N * 64), TB, 0, stream>>>(agg1, b1, hh, N);

    // ---- Layer 2 ----
    k_init2<<<blk(N), TB, 0, stream>>>(amax, denom, N);
    k_gemm2<<<(N + 15) / 16, 64, 0, stream>>>(hh, Wl2t, Wr2t, xl2, xr2, N);
    k_edge2_logits<<<blk(E), TB, 0, stream>>>(ei, xl2, xr2, att2, elog, amax, E);
    k_edge2_expsum<<<blk(E), TB, 0, stream>>>(ei, elog, amax, denom, E);
    k_edge2_agg<<<blk((long long)E * 4), TB, 0, stream>>>(ei, xl2, elog, denom, out, E);
    k_final_bias<<<blk((long long)N * 16), TB, 0, stream>>>(out, b2, N);
}
